// OldMultiHeadAttention_63763084476745
// MI455X (gfx1250) — matrix operand order repaired and checked by reading, not compiled, not run on hardware
//
#include <hip/hip_runtime.h>
#include <hip/hip_bf16.h>
#include <cstdint>

typedef __bf16 bf16;
typedef __attribute__((ext_vector_type(16))) __bf16 v16bf;
typedef __attribute__((ext_vector_type(8)))  __bf16 v8bf;
typedef __attribute__((ext_vector_type(2)))  __bf16 v2bf;
typedef __attribute__((ext_vector_type(8)))  float  v8f;

#define S_LEN  2048
#define CIN    256
#define DMODEL 256
#define NH     8
#define CH     32
#define BATCH  4
#define NTOK   (BATCH * S_LEN)   // 8192

// ---------------------------------------------------------------------------
// CDNA5 async global->LDS copy (ASYNCcnt path), with safe fallback.
#if __has_builtin(__builtin_amdgcn_global_load_async_to_lds_b128) && \
    __has_builtin(__builtin_amdgcn_s_wait_asynccnt)
#define HAVE_ASYNC_LDS 1
#endif

// The builtin's pointee type is a GCC-style int4 vector (per clang diagnostic).
typedef int b128_t __attribute__((vector_size(16)));
typedef __attribute__((address_space(1))) b128_t gb128_t;
typedef __attribute__((address_space(3))) b128_t sb128_t;

__device__ __forceinline__ void async_copy16(void* lds, const void* g) {
#ifdef HAVE_ASYNC_LDS
  // generic->as(3): LDS generic addresses carry the LDS offset in the low 32b.
  __builtin_amdgcn_global_load_async_to_lds_b128(
      (gb128_t*)(uintptr_t)g,
      (sb128_t*)(unsigned)(uintptr_t)lds,
      0, 0);
#else
  *(v8bf*)lds = *(const v8bf*)g;
#endif
}
__device__ __forceinline__ void async_wait0() {
#ifdef HAVE_ASYNC_LDS
  __builtin_amdgcn_s_wait_asynccnt(0);
#endif
}

// ---------------------------------------------------------------------------
// VALU-only 16-lane butterfly reductions (no ds_bpermute / dscnt waits).
#if __has_builtin(__builtin_amdgcn_mov_dpp) && __has_builtin(__builtin_amdgcn_permlane16)
#define HAVE_LANE_OPS 1
#endif

#ifdef HAVE_LANE_OPS
template <int CTRL>
__device__ __forceinline__ float dpp_mov(float v) {
  return __builtin_bit_cast(
      float, __builtin_amdgcn_mov_dpp(__builtin_bit_cast(int, v), CTRL, 0xF, 0xF, true));
}
__device__ __forceinline__ float permlane16_f(float v, unsigned lo, unsigned hi) {
  int i = __builtin_bit_cast(int, v);
  return __builtin_bit_cast(float, __builtin_amdgcn_permlane16(i, i, lo, hi, false, false));
}
#endif

__device__ __forceinline__ float red_max16(float v) {
#ifdef HAVE_LANE_OPS
  v = fmaxf(v, dpp_mov<0xB1>(v));                           // xor 1: quad_perm(1,0,3,2)
  v = fmaxf(v, dpp_mov<0x4E>(v));                           // xor 2: quad_perm(2,3,0,1)
  v = fmaxf(v, permlane16_f(v, 0x32107654u, 0xBA98FEDCu));  // xor 4
  v = fmaxf(v, permlane16_f(v, 0xFEDCBA98u, 0x76543210u));  // xor 8
#else
  v = fmaxf(v, __shfl_xor(v, 1, 32));
  v = fmaxf(v, __shfl_xor(v, 2, 32));
  v = fmaxf(v, __shfl_xor(v, 4, 32));
  v = fmaxf(v, __shfl_xor(v, 8, 32));
#endif
  return v;
}
__device__ __forceinline__ float red_sum16(float v) {
#ifdef HAVE_LANE_OPS
  v += dpp_mov<0xB1>(v);
  v += dpp_mov<0x4E>(v);
  v += permlane16_f(v, 0x32107654u, 0xBA98FEDCu);
  v += permlane16_f(v, 0xFEDCBA98u, 0x76543210u);
#else
  v += __shfl_xor(v, 1, 32);
  v += __shfl_xor(v, 2, 32);
  v += __shfl_xor(v, 4, 32);
  v += __shfl_xor(v, 8, 32);
#endif
  return v;
}

// ---------------------------------------------------------------------------
// WMMA fragment loaders (wave32, 16x16x32 bf16).
// A 16x32 (MxK): lane L -> m=L&15, h=L>>4; K = h*16..h*16+15 of row m
// => one contiguous 32B chunk of a row-major row (matches load_bfrag).
__device__ __forceinline__ v16bf load_afrag(const bf16* row, int h) {
  return *(const v16bf*)(row + h * 16);
}
// B 32x16 (KxN): lane L -> n=L&15, h=L>>4; K = h*16..h*16+15 of column n.
// With the matrix K-transposed in LDS (Bt[n][k]) this is ONE 32B chunk.
__device__ __forceinline__ v16bf load_bfrag(const bf16* p) {
  return *(const v16bf*)p;
}

// ---------------------------------------------------------------------------
// Kernel 1: input projections.
// MODE 0: q (scale 1/sqrt(32)) -> bf16 [B,H,S,C]
// MODE 1: k                     -> bf16 [B,H,S,C]
// MODE 2: v                     -> bf16 [B,H,C,S]  (pre-transposed for PV)
// MODE 3: g = sigmoid(x@Wg+bg)  -> f32  [B,S,D]
template <int MODE>
__global__ __launch_bounds__(256) void proj_kernel(
    const float* __restrict__ x, const float* __restrict__ W,
    const float* __restrict__ bvec, bf16* __restrict__ outb,
    float* __restrict__ outf) {
  __shared__ __align__(32) bf16 As[64][32];
  __shared__ __align__(32) bf16 Bs[64][32];

  const int tid = threadIdx.x;
  const int wid = tid >> 5, lane = tid & 31;
  const int ln = lane & 15, lh = lane >> 4;
  const int qw = wid & 3;   // M sub-tile (4 x 16 rows)
  const int nw = wid >> 2;  // N sub-tile (2 x 32 cols)
  const int m0 = blockIdx.x * 64;
  const int n0 = blockIdx.y * 64;

  v8f acc0 = {}, acc1 = {};

  for (int k0 = 0; k0 < CIN; k0 += 32) {
    __syncthreads();
    {  // x tile 64x32: f32 -> packed v8bf -> one ds_store_b128
      int r = tid >> 2, c = (tid & 3) * 8;
      const float* src = x + (size_t)(m0 + r) * CIN + k0 + c;
      v8bf t;
      #pragma unroll
      for (int j = 0; j < 8; ++j) t[j] = (bf16)src[j];
      *(v8bf*)&As[r][c] = t;
    }
    {  // W tile K-transposed: bf16 pairs -> ds_store_b32
      #pragma unroll
      for (int i = 0; i < 4; ++i) {
        int idx = tid + i * 256;
        int kp = idx >> 6, nn = idx & 63;  // kp: K pair index 0..15
        v2bf t;
        t[0] = (bf16)W[(size_t)(k0 + 2 * kp) * DMODEL + n0 + nn];
        t[1] = (bf16)W[(size_t)(k0 + 2 * kp + 1) * DMODEL + n0 + nn];
        *(v2bf*)&Bs[nn][2 * kp] = t;
      }
    }
    __syncthreads();
    v16bf a  = load_afrag(&As[qw * 16 + ln][0], lh);
    v16bf b0 = load_bfrag(&Bs[nw * 32 + ln][lh * 16]);
    v16bf b1 = load_bfrag(&Bs[nw * 32 + 16 + ln][lh * 16]);
    acc0 = __builtin_amdgcn_wmma_f32_16x16x32_bf16(false, a, false, b0,
                                                   (short)0, acc0, false, false);
    acc1 = __builtin_amdgcn_wmma_f32_16x16x32_bf16(false, a, false, b1,
                                                   (short)0, acc1, false, false);
  }

  if constexpr (MODE == 2) {
    // stage [n][token] in LDS, then coalesced transposed stores -> [B,H,C,S]
    __shared__ __align__(32) bf16 Stg[64][64];
    #pragma unroll
    for (int ct = 0; ct < 2; ++ct) {
      v8f acc = ct ? acc1 : acc0;
      v8bf pk;
      #pragma unroll
      for (int r = 0; r < 8; ++r) pk[r] = (bf16)acc[r];  // tokens qw*16+8lh+r
      *(v8bf*)&Stg[nw * 32 + ct * 16 + ln][qw * 16 + lh * 8] = pk;
    }
    __syncthreads();
    int n = tid >> 2, c16 = (tid & 3) * 16;
    int head = (n0 + n) >> 5, c = (n0 + n) & 31;
    int bb = m0 >> 11, s0 = m0 & (S_LEN - 1);
    v16bf val = *(const v16bf*)&Stg[n][c16];
    *(v16bf*)(outb + (((size_t)(bb * NH + head) * CH + c) * S_LEN + s0 + c16)) = val;
  } else {
    #pragma unroll
    for (int ct = 0; ct < 2; ++ct) {
      v8f acc = ct ? acc1 : acc0;
      #pragma unroll
      for (int r = 0; r < 8; ++r) {
        int m = m0 + qw * 16 + r + 8 * lh;       // token (b*S+s)
        int n = n0 + nw * 32 + ct * 16 + ln;     // model dim
        float v = acc[r];
        if constexpr (MODE == 3) {
          v += bvec[n];
          v = __builtin_amdgcn_rcpf(1.0f + __expf(-v));
          outf[(size_t)m * DMODEL + n] = v;
        } else {
          if constexpr (MODE == 0) v *= 0.17677669529663687f;  // 1/sqrt(32)
          int head = n >> 5, c = n & 31;
          int bb = m >> 11, s = m & (S_LEN - 1);
          outb[(((size_t)(bb * NH + head) * S_LEN) + s) * CH + c] = (bf16)v;
        }
      }
    }
  }
}

// ---------------------------------------------------------------------------
// Kernel 2: flash attention + gating, double-buffered async K/V staging.
// grid.x = S/128 (query blocks), grid.y = B*H. 8 waves, 16 queries each.
__global__ __launch_bounds__(256) void attn_kernel(
    const bf16* __restrict__ qbuf, const bf16* __restrict__ kbuf,
    const bf16* __restrict__ vtbuf, const float* __restrict__ gbuf,
    const float* __restrict__ bias, bf16* __restrict__ obuf) {
  __shared__ __align__(32) bf16 Qs[128][32];
  __shared__ __align__(32) bf16 Ks[2][64][32];   // [key][c]
  __shared__ __align__(32) bf16 Vts[2][32][64];  // [c][key]
  __shared__ __align__(32) bf16 Ps[8][16][64];   // per-wave P scratch

  const int tid = threadIdx.x, wid = tid >> 5, lane = tid & 31;
  const int ln = lane & 15, lh = lane >> 4;
  const int bh = blockIdx.y;  // b*8 + h
  const int head = bh & 7, bb = bh >> 3;
  const int q0 = blockIdx.x * 128;

  const bf16* Qg  = qbuf  + (size_t)bh * S_LEN * CH;
  const bf16* Kg  = kbuf  + (size_t)bh * S_LEN * CH;
  const bf16* Vtg = vtbuf + (size_t)bh * CH * S_LEN;
  const float* biasH = bias + (size_t)head * S_LEN * S_LEN;

  // staging helpers (all 256 threads; one async b128 per thread per tile)
  auto stage_kv = [&](int kb, int buf) {
    int r = tid >> 2, c = (tid & 3) * 8;
    async_copy16(&Ks[buf][r][c], Kg + (size_t)(kb + r) * CH + c);
    int cc = tid >> 3, k = (tid & 7) * 8;
    async_copy16(&Vts[buf][cc][k], Vtg + (size_t)cc * S_LEN + kb + k);
  };
  {  // Q tile 128x32 (two b128 chunks per thread)
    int r = tid >> 1, c = (tid & 1) * 16;
    const bf16* src = Qg + (size_t)(q0 + r) * CH + c;
    async_copy16(&Qs[r][c], src);
    async_copy16(&Qs[r][c + 8], src + 8);
  }
  stage_kv(0, 0);
  async_wait0();
  __syncthreads();
  const v16bf aq = load_afrag(&Qs[wid * 16 + ln][0], lh);

  v8f o0 = {}, o1 = {};
  float mrow[8], lrow[8];
  #pragma unroll
  for (int r = 0; r < 8; ++r) { mrow[r] = -1e30f; lrow[r] = 0.0f; }

  constexpr int NKT = S_LEN / 64;
  for (int kt = 0; kt < NKT; ++kt) {
    const int kb = kt * 64, cur = kt & 1;
    if (kt > 0) {  // wait for this tile's async data, then publish to all waves
      async_wait0();
      __syncthreads();
    }
    if (kt + 1 < NKT) stage_kv(kb + 64, cur ^ 1);  // prefetch next tile
    if (kt + 1 < NKT)
      __builtin_prefetch(&biasH[(size_t)(q0 + wid * 16 + lh * 8) * S_LEN + kb + 64], 0, 1);

    // S = Q @ K^T + bias  (4 x 16x16 tiles over 64 keys)
    v8f s[4];
    #pragma unroll
    for (int t = 0; t < 4; ++t) {
      v16bf bk = load_bfrag(&Ks[cur][t * 16 + ln][lh * 16]);
      v8f z = {};
      s[t] = __builtin_amdgcn_wmma_f32_16x16x32_bf16(false, aq, false, bk,
                                                     (short)0, z, false, false);
      #pragma unroll
      for (int r = 0; r < 8; ++r) {
        int q = q0 + wid * 16 + r + 8 * lh;
        s[t][r] += biasH[(size_t)q * S_LEN + kb + t * 16 + ln];
      }
    }

    // online softmax (row M = r + 8*lh lives within one 16-lane half);
    // reductions are pure VALU (DPP + permlane16), no LDS round-trips.
    float scl[8];
    #pragma unroll
    for (int r = 0; r < 8; ++r) {
      float mx = fmaxf(fmaxf(s[0][r], s[1][r]), fmaxf(s[2][r], s[3][r]));
      mx = red_max16(mx);
      float mnew = fmaxf(mrow[r], mx);
      scl[r] = __expf(mrow[r] - mnew);
      mrow[r] = mnew;
      float ls = 0.0f;
      #pragma unroll
      for (int t = 0; t < 4; ++t) {
        float p = __expf(s[t][r] - mnew);
        s[t][r] = p;
        ls += p;
      }
      ls = red_sum16(ls);
      lrow[r] = lrow[r] * scl[r] + ls;
    }

    // P (C/D layout) -> per-wave LDS scratch -> A-fragment layout
    #pragma unroll
    for (int t = 0; t < 4; ++t)
      #pragma unroll
      for (int r = 0; r < 8; ++r)
        Ps[wid][r + 8 * lh][t * 16 + ln] = (bf16)s[t][r];
    #pragma unroll
    for (int r = 0; r < 8; ++r) { o0[r] *= scl[r]; o1[r] *= scl[r]; }

    v16bf a0 = load_afrag(&Ps[wid][ln][0], lh);   // keys 0..31
    v16bf a1 = load_afrag(&Ps[wid][ln][32], lh);  // keys 32..63
    #pragma unroll
    for (int ct = 0; ct < 2; ++ct) {
      v16bf b0 = load_bfrag(&Vts[cur][ct * 16 + ln][lh * 16]);
      v16bf b1 = load_bfrag(&Vts[cur][ct * 16 + ln][32 + lh * 16]);
      v8f& oo = ct ? o1 : o0;
      oo = __builtin_amdgcn_wmma_f32_16x16x32_bf16(false, a0, false, b0,
                                                   (short)0, oo, false, false);
      oo = __builtin_amdgcn_wmma_f32_16x16x32_bf16(false, a1, false, b1,
                                                   (short)0, oo, false, false);
    }
  }

  // normalize, gate, store token-major bf16 [B,S,D]
  #pragma unroll
  for (int ct = 0; ct < 2; ++ct) {
    v8f oo = ct ? o1 : o0;
    #pragma unroll
    for (int r = 0; r < 8; ++r) {
      int q = q0 + wid * 16 + r + 8 * lh;
      size_t tok = (size_t)bb * S_LEN + q;
      int n = head * CH + ct * 16 + ln;
      float inv_l = __builtin_amdgcn_rcpf(lrow[r]);
      float val = oo[r] * inv_l * gbuf[tok * DMODEL + n];
      obuf[tok * DMODEL + n] = (bf16)val;
    }
  }
}

// ---------------------------------------------------------------------------
// Kernel 3: output projection  out = (g*o) @ Wo + bo  (f32 out)
__global__ __launch_bounds__(256) void outproj_kernel(
    const bf16* __restrict__ A, const float* __restrict__ W,
    const float* __restrict__ bo, float* __restrict__ out) {
  __shared__ __align__(32) bf16 As[64][32];
  __shared__ __align__(32) bf16 Bs[64][32];

  const int tid = threadIdx.x;
  const int wid = tid >> 5, lane = tid & 31;
  const int ln = lane & 15, lh = lane >> 4;
  const int qw = wid & 3, nw = wid >> 2;
  const int m0 = blockIdx.x * 64;
  const int n0 = blockIdx.y * 64;

  v8f acc0 = {}, acc1 = {};

  for (int k0 = 0; k0 < DMODEL; k0 += 32) {
    __syncthreads();
    {  // A already bf16: async b128 staging
      int r = tid >> 2, c = (tid & 3) * 8;
      async_copy16(&As[r][c], A + (size_t)(m0 + r) * DMODEL + k0 + c);
    }
    {  // Wo tile K-transposed, bf16 pairs
      #pragma unroll
      for (int i = 0; i < 4; ++i) {
        int idx = tid + i * 256;
        int kp = idx >> 6, nn = idx & 63;
        v2bf t;
        t[0] = (bf16)W[(size_t)(k0 + 2 * kp) * CIN + n0 + nn];
        t[1] = (bf16)W[(size_t)(k0 + 2 * kp + 1) * CIN + n0 + nn];
        *(v2bf*)&Bs[nn][2 * kp] = t;
      }
    }
    async_wait0();
    __syncthreads();
    v16bf a  = load_afrag(&As[qw * 16 + ln][0], lh);
    v16bf b0 = load_bfrag(&Bs[nw * 32 + ln][lh * 16]);
    v16bf b1 = load_bfrag(&Bs[nw * 32 + 16 + ln][lh * 16]);
    acc0 = __builtin_amdgcn_wmma_f32_16x16x32_bf16(false, a, false, b0,
                                                   (short)0, acc0, false, false);
    acc1 = __builtin_amdgcn_wmma_f32_16x16x32_bf16(false, a, false, b1,
                                                   (short)0, acc1, false, false);
  }

  #pragma unroll
  for (int ct = 0; ct < 2; ++ct) {
    v8f acc = ct ? acc1 : acc0;
    #pragma unroll
    for (int r = 0; r < 8; ++r) {
      int m = m0 + qw * 16 + r + 8 * lh;
      int n = n0 + nw * 32 + ct * 16 + ln;
      out[(size_t)m * CIN + n] = acc[r] + bo[n];
    }
  }
}

// ---------------------------------------------------------------------------
extern "C" void kernel_launch(void* const* d_in, const int* in_sizes, int n_in,
                              void* d_out, int out_size, void* d_ws, size_t ws_size,
                              hipStream_t stream) {
  const float* x    = (const float*)d_in[0];
  const float* bias = (const float*)d_in[1];
  const float* Wq   = (const float*)d_in[2];
  const float* Wk   = (const float*)d_in[3];
  const float* Wv   = (const float*)d_in[4];
  const float* Wg   = (const float*)d_in[5];
  const float* bg   = (const float*)d_in[6];
  const float* Wo   = (const float*)d_in[7];
  const float* bo   = (const float*)d_in[8];
  float* out = (float*)d_out;

  char* ws = (char*)d_ws;
  bf16*  qb = (bf16*)(ws);                       // 4 MB  [B,H,S,C] bf16
  bf16*  kb = (bf16*)(ws + ((size_t)4  << 20));  // 4 MB  [B,H,S,C] bf16
  bf16*  vb = (bf16*)(ws + ((size_t)8  << 20));  // 4 MB  [B,H,C,S] bf16 (transposed)
  float* gb = (float*)(ws + ((size_t)12 << 20)); // 8 MB  [B,S,D] f32
  bf16*  ob = (bf16*)(ws + ((size_t)20 << 20));  // 4 MB  [B,S,D] bf16

  dim3 blk(256);
  dim3 g1(NTOK / 64, DMODEL / 64);  // 128 x 4
  proj_kernel<0><<<g1, blk, 0, stream>>>(x, Wq, bg, qb, gb);
  proj_kernel<1><<<g1, blk, 0, stream>>>(x, Wk, bg, kb, gb);
  proj_kernel<2><<<g1, blk, 0, stream>>>(x, Wv, bg, vb, gb);
  proj_kernel<3><<<g1, blk, 0, stream>>>(x, Wg, bg, qb, gb);

  dim3 g2(S_LEN / 128, BATCH * NH);  // 16 x 32
  attn_kernel<<<g2, blk, 0, stream>>>(qb, kb, vb, gb, bias, ob);

  dim3 g3(NTOK / 64, CIN / 64);  // 128 x 4
  outproj_kernel<<<g3, blk, 0, stream>>>(ob, Wo, bo, out);
}
